// SelfAttentionLayer_30923764531456
// MI455X (gfx1250) — matrix operand order repaired and checked by reading, not compiled, not run on hardware
//
#include <hip/hip_runtime.h>
#include <hip/hip_bf16.h>
#include <stdint.h>

// ---------------------------------------------------------------------------
// Self-attention for MI455X (gfx1250): bf16 WMMA + async global->LDS staging.
// B=4, N=4096, D=1024, H=128, causal, tau = sqrt(H).
// ---------------------------------------------------------------------------

#define B_  4
#define N_  4096
#define D_  1024
#define H_  128
#define MT_ (B_ * N_)   // 16384 total rows

typedef __attribute__((ext_vector_type(16))) __bf16 v16bf;
typedef __attribute__((ext_vector_type(8)))  float  v8f;

union FragBf16 {                 // one WMMA A/B operand: 16 bf16 x 32 lanes
    v16bf v;
    uint4 q[2];
};

static __device__ __forceinline__ unsigned short f2bf(float f) {
    union { float f; unsigned u; } c; c.f = f;
    unsigned r = c.u + 0x7FFFu + ((c.u >> 16) & 1u);   // round-to-nearest-even
    return (unsigned short)(r >> 16);
}

static __device__ __forceinline__ v8f wmma_bf16(v16bf a, v16bf b, v8f c) {
    return __builtin_amdgcn_wmma_f32_16x16x32_bf16(
        /*neg_a=*/false, a, /*neg_b=*/false, b,
        /*c_mod=*/(short)0, c, /*reuse_a=*/false, /*reuse_b=*/false);
}

// ---- CDNA5 async global->LDS copy (ASYNCcnt), with sync fallback ----------
#if __has_builtin(__builtin_amdgcn_global_load_async_to_lds_b128)
#define HAS_ASYNC_LDS 1
#else
#define HAS_ASYNC_LDS 0
#endif

#define AS1 __attribute__((address_space(1)))
#define AS3 __attribute__((address_space(3)))

typedef __attribute__((__vector_size__(4 * sizeof(int)))) int v4i_;

// copy 16 bytes (8 bf16) per lane, global -> LDS
static __device__ __forceinline__ void acp16(const unsigned short* g,
                                             unsigned short* l) {
#if HAS_ASYNC_LDS
    __builtin_amdgcn_global_load_async_to_lds_b128(
        (AS1 v4i_*)g, (AS3 v4i_*)l, 0, 0);
#else
    *(uint4*)l = *(const uint4*)g;
#endif
}

template <int N>
static __device__ __forceinline__ void wait_async() {
#if HAS_ASYNC_LDS
#if __has_builtin(__builtin_amdgcn_s_wait_asynccnt)
    __builtin_amdgcn_s_wait_asynccnt(N);
#else
    asm volatile("s_wait_asynccnt %0" ::"i"(N) : "memory");
#endif
#endif
}

// ---------------------------------------------------------------------------
// fp32 -> bf16 elementwise (x and all weights; no transpose needed)
// ---------------------------------------------------------------------------
__global__ __launch_bounds__(256) void k_cvt(const float* __restrict__ src,
                                             unsigned short* __restrict__ dst,
                                             int n) {
    int i = (blockIdx.x * 256 + threadIdx.x) * 4;
    if (i < n) {
        float4 f = *(const float4*)(src + i);
        uint2 o;
        o.x = (unsigned)f2bf(f.x) | ((unsigned)f2bf(f.y) << 16);
        o.y = (unsigned)f2bf(f.z) | ((unsigned)f2bf(f.w) << 16);
        *(uint2*)(dst + i) = o;
    }
}

// ---------------------------------------------------------------------------
// bf16 GEMM: C = A[M,K] * W[Nout,K]^T, f32 accum, double-buffered async LDS.
// Block 128x64, 8 waves as 4(M)x2(N), 32x32 per wave, K-step 32.
// STORE_T: store C transposed per batch as [B][Nout][N_] (for v^T).
// ---------------------------------------------------------------------------
template <bool STORE_T>
__global__ __launch_bounds__(256) void k_gemm_bf16(
    const unsigned short* __restrict__ A,   // [M,K] bf16
    const unsigned short* __restrict__ W,   // [Nout,K] bf16 (nn.Linear layout)
    unsigned short* __restrict__ C,
    int M, int K, int Nout) {
    __shared__ __align__(16) unsigned short Al[2][128 * 32];
    __shared__ __align__(16) unsigned short Bl[2][64 * 32];   // [n][k]

    const int tid  = threadIdx.x;
    const int lane = tid & 31, wave = tid >> 5;
    const int rg = wave >> 1, cg = wave & 1;
    const int nlo = lane & 15, hi = lane >> 4;
    const int m0 = blockIdx.x * 128;
    const int n0 = blockIdx.y * 64;

    auto stage = [&](int k0, int bufi) {
        {   // A tile 128x32: 2 async b128 per thread
            int r = tid >> 1, c = (tid & 1) * 16;
            const unsigned short* g = A + (size_t)(m0 + r) * K + k0 + c;
            unsigned short* l = Al[bufi] + r * 32 + c;
            acp16(g, l);
            acp16(g + 8, l + 8);
        }
        {   // B tile: W rows n0..n0+63, cols k0..k0+31 -> LDS [n][k] directly
            int n = tid >> 2, c = (tid & 3) * 8;
            acp16(W + (size_t)(n0 + n) * K + k0 + c, Bl[bufi] + n * 32 + c);
        }
    };

    v8f acc[2][2];
#pragma unroll
    for (int ti = 0; ti < 2; ti++)
#pragma unroll
        for (int tj = 0; tj < 2; tj++)
#pragma unroll
            for (int e = 0; e < 8; e++) acc[ti][tj][e] = 0.0f;

    stage(0, 0);
    const int nk = K / 32;
    for (int i = 0; i < nk; i++) {
        if (i + 1 < nk) {
            stage((i + 1) * 32, (i + 1) & 1);
            wait_async<3>();            // tile i done, tile i+1 in flight
        } else {
            wait_async<0>();
        }
        __syncthreads();
        const unsigned short* Ac = Al[i & 1];
        const unsigned short* Bc = Bl[i & 1];
#pragma unroll
        for (int ti = 0; ti < 2; ti++) {
            FragBf16 af;
            const unsigned short* pa = Ac + (rg * 32 + ti * 16 + nlo) * 32 + hi * 8;
            af.q[0] = *(const uint4*)pa;
            af.q[1] = *(const uint4*)(pa + 16);
#pragma unroll
            for (int tj = 0; tj < 2; tj++) {
                FragBf16 bf;
                const unsigned short* pb = Bc + (cg * 32 + tj * 16 + nlo) * 32 + hi * 8;
                bf.q[0] = *(const uint4*)pb;
                bf.q[1] = *(const uint4*)(pb + 16);
                acc[ti][tj] = wmma_bf16(af.v, bf.v, acc[ti][tj]);
            }
        }
        __syncthreads();
    }

#pragma unroll
    for (int ti = 0; ti < 2; ti++)
#pragma unroll
        for (int tj = 0; tj < 2; tj++)
#pragma unroll
            for (int i = 0; i < 8; i++) {
                int row = m0 + rg * 32 + ti * 16 + i + 8 * hi;
                int col = n0 + cg * 32 + tj * 16 + nlo;
                unsigned short val = f2bf(acc[ti][tj][i]);
                if (STORE_T) {
                    int bb = row >> 12;            // row / N_
                    int n  = row & (N_ - 1);       // row % N_
                    C[(((size_t)bb * Nout + col) << 12) + n] = val;
                } else {
                    C[(size_t)row * Nout + col] = val;
                }
            }
}

// ---------------------------------------------------------------------------
// Causal flash attention. Grid: (N/64, D/256, B), 256 threads (8 waves).
// Wave = 16 q-rows x 128 out-cols (8 accum tiles). Key tiles of 32,
// double-buffered async K/V staging; V comes pre-transposed [B][D][N].
// ---------------------------------------------------------------------------
__global__ __launch_bounds__(256) void k_flash(
    const unsigned short* __restrict__ Q,   // [B,N,H]  bf16
    const unsigned short* __restrict__ Kb,  // [B,N,H]  bf16
    const unsigned short* __restrict__ Vt,  // [B,D,N]  bf16 (v transposed)
    float* __restrict__ O, float invtau) {
    union SMemU {                           // Q staging aliases K/V buffers
        unsigned short Ql[64 * 128];        // 16 KB, prologue only
        struct {
            unsigned short Kl[2][32 * 128]; // 16 KB
            unsigned short Vl[2][256 * 32]; // 32 KB  [col][key]
            unsigned short Pl[8 * 16 * 32]; //  8 KB
        } s;
    };
    __shared__ __align__(16) SMemU sm;

    const int tid  = threadIdx.x;
    const int lane = tid & 31, wave = tid >> 5;
    const int rg = wave >> 1, cg = wave & 1;
    const int nlo = lane & 15, hi = lane >> 4;
    const int q0 = blockIdx.x * 64;
    const int c0 = blockIdx.y * 256;
    const int b  = blockIdx.z;

    // ---- prologue: stage Q (64x128), load per-wave Q fragments ----
    {
        int r = tid >> 2, cb = (tid & 3) * 32;
        const unsigned short* g = Q + ((size_t)b * N_ + q0 + r) * H_ + cb;
        unsigned short* l = sm.Ql + r * 128 + cb;
        acp16(g, l); acp16(g + 8, l + 8);
        acp16(g + 16, l + 16); acp16(g + 24, l + 24);
    }
    wait_async<0>();
    __syncthreads();
    FragBf16 qf[4];
#pragma unroll
    for (int hc = 0; hc < 4; hc++) {
        const unsigned short* p = sm.Ql + (rg * 16 + nlo) * 128 + hc * 32 + hi * 8;
        qf[hc].q[0] = *(const uint4*)p;
        qf[hc].q[1] = *(const uint4*)(p + 16);
    }
    asm volatile("s_wait_dscnt 0" ::: "memory");
    __syncthreads();                        // Ql dead; union space reused below

    auto stageKV = [&](int kt2, int bufi) {
        const int ky = kt2 * 32;
        {   // K tile 32x128
            int r = tid >> 3, cb = (tid & 7) * 16;
            const unsigned short* g = Kb + ((size_t)b * N_ + ky + r) * H_ + cb;
            unsigned short* l = sm.s.Kl[bufi] + r * 128 + cb;
            acp16(g, l); acp16(g + 8, l + 8);
        }
        {   // V tile: 256 cols x 32 keys, contiguous rows of v^T
            const unsigned short* g = Vt + ((size_t)b * D_ + c0 + tid) * N_ + ky;
            unsigned short* l = sm.s.Vl[bufi] + tid * 32;
            acp16(g, l); acp16(g + 8, l + 8);
            acp16(g + 16, l + 16); acp16(g + 24, l + 24);
        }
    };

    v8f Oa[8];
    float m[8], lsum[8];
#pragma unroll
    for (int j = 0; j < 8; j++)
#pragma unroll
        for (int e = 0; e < 8; e++) Oa[j][e] = 0.0f;
#pragma unroll
    for (int i = 0; i < 8; i++) { m[i] = -__builtin_inff(); lsum[i] = 0.0f; }

    unsigned short* ps = sm.s.Pl + wave * 512;   // wave-private 16x32 P

    const int nkt = (q0 >> 5) + 2;          // key tiles with ky <= q0+63
    stageKV(0, 0);
    for (int kt = 0; kt < nkt; kt++) {
        const int ky0 = kt * 32;
        if (kt + 1 < nkt) {
            stageKV(kt + 1, (kt + 1) & 1);
            wait_async<6>();                // tile kt done, kt+1 in flight
        } else {
            wait_async<0>();
        }
        __syncthreads();
        const unsigned short* Klc = sm.s.Kl[kt & 1];
        const unsigned short* Vlc = sm.s.Vl[kt & 1];

        // ---- scores S[16 x 32] = Q(16xH) * K^T(Hx32), f32 accum ----
        v8f S0, S1;
#pragma unroll
        for (int e = 0; e < 8; e++) { S0[e] = 0.0f; S1[e] = 0.0f; }
#pragma unroll
        for (int hc = 0; hc < 4; hc++) {
            FragBf16 k0f, k1f;
            const unsigned short* b0 = Klc + nlo * 128 + hc * 32 + hi * 8;
            k0f.q[0] = *(const uint4*)b0;
            k0f.q[1] = *(const uint4*)(b0 + 16);
            const unsigned short* b1 = Klc + (16 + nlo) * 128 + hc * 32 + hi * 8;
            k1f.q[0] = *(const uint4*)b1;
            k1f.q[1] = *(const uint4*)(b1 + 16);
            S0 = wmma_bf16(qf[hc].v, k0f.v, S0);
            S1 = wmma_bf16(qf[hc].v, k1f.v, S1);
        }

        // ---- online softmax (VGPR i <-> row i + 8*hi) ----
#pragma unroll
        for (int i = 0; i < 8; i++) {
            int row_g = q0 + rg * 16 + i + 8 * hi;
            int cg0 = ky0 + nlo, cg1 = cg0 + 16;
            float s0 = S0[i] * invtau; if (cg0 > row_g) s0 = -__builtin_inff();
            float s1 = S1[i] * invtau; if (cg1 > row_g) s1 = -__builtin_inff();
            float t = fmaxf(s0, s1);
#pragma unroll
            for (int off = 1; off < 16; off <<= 1)
                t = fmaxf(t, __shfl_xor(t, off, 32));
            float mn = fmaxf(m[i], t);
            float alpha = __expf(m[i] - mn);
            float p0 = __expf(s0 - mn), p1 = __expf(s1 - mn);
            float rs = p0 + p1;
#pragma unroll
            for (int off = 1; off < 16; off <<= 1)
                rs += __shfl_xor(rs, off, 32);
            lsum[i] = lsum[i] * alpha + rs;
            m[i] = mn;
#pragma unroll
            for (int j = 0; j < 8; j++) Oa[j][i] *= alpha;
            int rl = i + 8 * hi;
            ps[rl * 32 + nlo]      = f2bf(p0);
            ps[rl * 32 + nlo + 16] = f2bf(p1);
        }
        asm volatile("s_wait_dscnt 0" ::: "memory");  // wave-local P restage

        // ---- O += P(16x32) * V(32x128 slice) ----
        FragBf16 pf;
        {
            const unsigned short* bp = ps + nlo * 32 + hi * 8;
            pf.q[0] = *(const uint4*)bp;
            pf.q[1] = *(const uint4*)(bp + 16);
        }
#pragma unroll
        for (int j = 0; j < 8; j++) {
            FragBf16 vf;
            const unsigned short* bv = Vlc + (cg * 128 + j * 16 + nlo) * 32 + hi * 8;
            vf.q[0] = *(const uint4*)bv;
            vf.q[1] = *(const uint4*)(bv + 16);
            Oa[j] = wmma_bf16(pf.v, vf.v, Oa[j]);
        }
        __syncthreads();
    }

    // ---- epilogue: normalize and write fp32 output ----
#pragma unroll
    for (int i = 0; i < 8; i++) {
        float inv = 1.0f / lsum[i];
        int row = q0 + rg * 16 + i + 8 * hi;
        float* og = O + ((size_t)b * N_ + row) * D_ + c0 + cg * 128 + nlo;
#pragma unroll
        for (int j = 0; j < 8; j++) og[j * 16] = Oa[j][i] * inv;
    }
}

// ---------------------------------------------------------------------------
extern "C" void kernel_launch(void* const* d_in, const int* in_sizes, int n_in,
                              void* d_out, int out_size, void* d_ws,
                              size_t ws_size, hipStream_t stream) {
    (void)in_sizes; (void)n_in; (void)out_size; (void)ws_size;
    const float* x  = (const float*)d_in[0];
    const float* Wq = (const float*)d_in[1];
    const float* Wk = (const float*)d_in[2];
    const float* Wv = (const float*)d_in[3];
    float* out = (float*)d_out;

    char* ws = (char*)d_ws;
    const size_t szXB = (size_t)MT_ * D_ * 2;   // x bf16        32 MB
    const size_t szWQ = (size_t)H_ * D_ * 2;    // Wq bf16       256 KB
    const size_t szWV = (size_t)D_ * D_ * 2;    // Wv bf16       2 MB
    const size_t szQ  = (size_t)MT_ * H_ * 2;   // q bf16        4 MB

    unsigned short* xb  = (unsigned short*)(ws);
    unsigned short* wqb = (unsigned short*)(ws + szXB);
    unsigned short* wkb = (unsigned short*)(ws + szXB + szWQ);
    unsigned short* wvb = (unsigned short*)(ws + szXB + 2 * szWQ);
    unsigned short* qb  = (unsigned short*)(ws + szXB + 2 * szWQ + szWV);
    unsigned short* kb  = (unsigned short*)(ws + szXB + 2 * szWQ + szWV + szQ);
    unsigned short* vT  = (unsigned short*)(ws + szXB + 2 * szWQ + szWV + 2 * szQ);

    // 1) convert inputs to bf16 (layouts preserved)
    k_cvt<<<(MT_ * D_) / (256 * 4), 256, 0, stream>>>(x, xb, MT_ * D_);
    k_cvt<<<(H_ * D_) / (256 * 4), 256, 0, stream>>>(Wq, wqb, H_ * D_);
    k_cvt<<<(H_ * D_) / (256 * 4), 256, 0, stream>>>(Wk, wkb, H_ * D_);
    k_cvt<<<(D_ * D_) / (256 * 4), 256, 0, stream>>>(Wv, wvb, D_ * D_);

    // 2) projections (bf16 WMMA GEMMs); v stored transposed for attention
    k_gemm_bf16<false><<<dim3(MT_ / 128, H_ / 64), 256, 0, stream>>>(
        xb, wqb, qb, MT_, D_, H_);
    k_gemm_bf16<false><<<dim3(MT_ / 128, H_ / 64), 256, 0, stream>>>(
        xb, wkb, kb, MT_, D_, H_);
    k_gemm_bf16<true><<<dim3(MT_ / 128, D_ / 64), 256, 0, stream>>>(
        xb, wvb, vT, MT_, D_, D_);

    // 3) causal flash attention
    const float invtau = 0.08838834764831845f;  // 1/sqrt(128)
    k_flash<<<dim3(N_ / 64, D_ / 256, B_), 256, 0, stream>>>(
        qb, kb, vT, out, invtau);
}
